// LongformerSelfAttention_65687229825616
// MI455X (gfx1250) — matrix operand order repaired and checked by reading, not compiled, not run on hardware
//
#include <hip/hip_runtime.h>
#include <hip/hip_bf16.h>

typedef __attribute__((ext_vector_type(16))) _Float16 v16h;
typedef __attribute__((ext_vector_type(8)))  float    v8f;
typedef _Float16 half_t;

#define DM    1024      // d_model
#define HEADS 16
#define HD    64        // head dim
#define TSEQ  2048
#define BATCH 2
#define NTOK  (BATCH * TSEQ)   // 4096
#define WIN   128

// ---------------------------------------------------------------------------
// WMMA operand loaders (contiguous k = 16*(l/16)+e for A and B, wave32)
// ---------------------------------------------------------------------------

// A-matrix 16x32 f16: lane l holds row (l&15); elements 0..7 -> K = base+e,
// elements 8..15 -> K = base+8+(e-8), base = (l<16)?0:16.  Two contiguous 16B runs.
__device__ __forceinline__ v16h load_a16x32(const half_t* src, int ld, int row0, int k0) {
    const int l    = threadIdx.x & 31;
    const int row  = row0 + (l & 15);
    const int base = (l < 16) ? 0 : 16;
    const half_t* p = src + (size_t)row * ld + k0;
    v16h a;
#pragma unroll
    for (int e = 0; e < 8; ++e) a[e]     = p[base + e];
#pragma unroll
    for (int e = 0; e < 8; ++e) a[8 + e] = p[base + 8 + e];
    return a;
}

// B-matrix 32x16 f16 where B(k,n) = src[(n0+n)*ld + k0 + k]  (n-major source,
// e.g. W[N][K] for Y = X*W^T, or K[token][feat] for Q*K^T).
// Lane l holds column n = (l&15); elements e -> K = (l<16?0:16)+e, contiguous.
__device__ __forceinline__ v16h load_b32x16(const half_t* src, int ld, int n0, int k0) {
    const int l  = threadIdx.x & 31;
    const int n  = n0 + (l & 15);
    const int kb = k0 + ((l < 16) ? 0 : 16);
    const half_t* p = src + (size_t)n * ld + kb;
    v16h b;
#pragma unroll
    for (int e = 0; e < 16; ++e) b[e] = p[e];
    return b;
}

__device__ __forceinline__ v8f wmma_f16(v16h a, v16h b, v8f c) {
    return __builtin_amdgcn_wmma_f32_16x16x32_f16(false, a, false, b, (short)0, c, false, false);
}

__device__ __forceinline__ float redmax16(float v) {
    v = fmaxf(v, __shfl_xor(v, 1, 32));
    v = fmaxf(v, __shfl_xor(v, 2, 32));
    v = fmaxf(v, __shfl_xor(v, 4, 32));
    v = fmaxf(v, __shfl_xor(v, 8, 32));
    return v;   // butterfly: all 16 lanes of the half-group hold the max
}
__device__ __forceinline__ float redsum16(float v) {
    v += __shfl_xor(v, 1, 32);
    v += __shfl_xor(v, 2, 32);
    v += __shfl_xor(v, 4, 32);
    v += __shfl_xor(v, 8, 32);
    return v;
}

// ---------------------------------------------------------------------------
// fp32 -> fp16 conversion
// ---------------------------------------------------------------------------
__global__ void f32_to_f16_kernel(const float* __restrict__ src,
                                  half_t* __restrict__ dst, int n) {
    int i = blockIdx.x * blockDim.x + threadIdx.x;
    if (i < n) dst[i] = (half_t)src[i];
}

// ---------------------------------------------------------------------------
// GEMM: Y = A(f16, MxK) * W(f16, NxK)^T + bias.
// Each wave computes a 32x32 C block (4 WMMA accumulators; 2 A + 2 B tiles
// per k-step -> ~16 flops/byte from cache).  Block = (32,4): 4 waves tile N,
// so a block covers M 32 x N 128.
// ---------------------------------------------------------------------------
__global__ __launch_bounds__(128)
void gemm_f16out(const half_t* __restrict__ A, const half_t* __restrict__ W,
                 const float* __restrict__ bias, half_t* __restrict__ Y,
                 int M, int N, int K, int transStore) {
    const int n0 = blockIdx.x * 128 + threadIdx.y * 32;
    const int m0 = blockIdx.y * 32;
    v8f c00 = {}, c01 = {}, c10 = {}, c11 = {};
    for (int k0 = 0; k0 < K; k0 += 32) {
        if (k0 + 128 < K) {   // software prefetch -> global_prefetch_b8
            __builtin_prefetch(A + (size_t)(m0 + (threadIdx.x & 31)) * K + k0 + 128, 0, 1);
            __builtin_prefetch(W + (size_t)(n0 + (threadIdx.x & 15)) * K + k0 + 128, 0, 1);
        }
        v16h a0 = load_a16x32(A, K, m0,      k0);
        v16h a1 = load_a16x32(A, K, m0 + 16, k0);
        v16h b0 = load_b32x16(W, K, n0,      k0);
        v16h b1 = load_b32x16(W, K, n0 + 16, k0);
        c00 = wmma_f16(a0, b0, c00);
        c01 = wmma_f16(a0, b1, c01);
        c10 = wmma_f16(a1, b0, c10);
        c11 = wmma_f16(a1, b1, c11);
    }
    const int l    = threadIdx.x & 31;
    const int col0 = n0 + (l & 15);
    const int row0 = m0 + ((l < 16) ? 0 : 8);
    const float bv0 = bias[col0], bv1 = bias[col0 + 16];
#pragma unroll
    for (int r = 0; r < 8; ++r) {
        float v00 = c00[r] + bv0;
        float v01 = c01[r] + bv1;
        float v10 = c10[r] + bv0;
        float v11 = c11[r] + bv1;
        if (transStore) {   // Y[n][m]  (for V^T layout)
            Y[(size_t)col0 * M + row0 + r]             = (half_t)v00;
            Y[(size_t)(col0 + 16) * M + row0 + r]      = (half_t)v01;
            Y[(size_t)col0 * M + row0 + 16 + r]        = (half_t)v10;
            Y[(size_t)(col0 + 16) * M + row0 + 16 + r] = (half_t)v11;
        } else {            // Y[m][n]
            Y[(size_t)(row0 + r) * N + col0]           = (half_t)v00;
            Y[(size_t)(row0 + r) * N + col0 + 16]      = (half_t)v01;
            Y[(size_t)(row0 + 16 + r) * N + col0]      = (half_t)v10;
            Y[(size_t)(row0 + 16 + r) * N + col0 + 16] = (half_t)v11;
        }
    }
}

// Same GEMM but fp32 output (final projection into d_out)
__global__ __launch_bounds__(128)
void gemm_f32out(const half_t* __restrict__ A, const half_t* __restrict__ W,
                 const float* __restrict__ bias, float* __restrict__ Y,
                 int M, int N, int K) {
    const int n0 = blockIdx.x * 128 + threadIdx.y * 32;
    const int m0 = blockIdx.y * 32;
    v8f c00 = {}, c01 = {}, c10 = {}, c11 = {};
    for (int k0 = 0; k0 < K; k0 += 32) {
        if (k0 + 128 < K) {
            __builtin_prefetch(A + (size_t)(m0 + (threadIdx.x & 31)) * K + k0 + 128, 0, 1);
            __builtin_prefetch(W + (size_t)(n0 + (threadIdx.x & 15)) * K + k0 + 128, 0, 1);
        }
        v16h a0 = load_a16x32(A, K, m0,      k0);
        v16h a1 = load_a16x32(A, K, m0 + 16, k0);
        v16h b0 = load_b32x16(W, K, n0,      k0);
        v16h b1 = load_b32x16(W, K, n0 + 16, k0);
        c00 = wmma_f16(a0, b0, c00);
        c01 = wmma_f16(a0, b1, c01);
        c10 = wmma_f16(a1, b0, c10);
        c11 = wmma_f16(a1, b1, c11);
    }
    const int l    = threadIdx.x & 31;
    const int col0 = n0 + (l & 15);
    const int row0 = m0 + ((l < 16) ? 0 : 8);
    const float bv0 = bias[col0], bv1 = bias[col0 + 16];
#pragma unroll
    for (int r = 0; r < 8; ++r) {
        Y[(size_t)(row0 + r) * N + col0]           = c00[r] + bv0;
        Y[(size_t)(row0 + r) * N + col0 + 16]      = c01[r] + bv1;
        Y[(size_t)(row0 + 16 + r) * N + col0]      = c10[r] + bv0;
        Y[(size_t)(row0 + 16 + r) * N + col0 + 16] = c11[r] + bv1;
    }
}

// ---------------------------------------------------------------------------
// Flash-style windowed attention. One wave per (b, h, 16-query tile).
//   Qh, Kh : f16 [NTOK][DM]   (head h occupies cols h*64 .. h*64+63)
//   Vt     : f16 [DM][NTOK]   (transposed: V feature-major for B-matrix use)
//   Att    : f16 [NTOK][DM]   attention output (pre output-projection)
// Mask: (c <= q && q-c <= WIN) || q==0 || c==0
// ---------------------------------------------------------------------------
__global__ __launch_bounds__(32)
void attn_kernel(const half_t* __restrict__ Qh, const half_t* __restrict__ Kh,
                 const half_t* __restrict__ Vt, half_t* __restrict__ Att) {
    const int qt = blockIdx.x;           // 0..127 query tile
    const int h  = blockIdx.y;           // head
    const int b  = blockIdx.z;           // batch
    const int q0 = qt * 16;
    const int l      = threadIdx.x & 31;
    const int hi8    = (l < 16) ? 0 : 8; // C-layout row offset
    const int colIdx = l & 15;           // C-layout column
    const int mrow0  = b * TSEQ;         // token offset of this batch

    __shared__ half_t pst[16 * 32];      // P staging: f32 C-layout -> f16 A-layout

    const half_t* Qbase = Qh + (size_t)mrow0 * DM + h * HD;
    const half_t* Kbase = Kh + (size_t)mrow0 * DM + h * HD;
    const half_t* Vbase = Vt + (size_t)(h * HD) * NTOK + mrow0;

    // Q A-tiles, resident for the whole tile (features 0..31 / 32..63)
    const v16h qa0 = load_a16x32(Qbase, DM, q0, 0);
    const v16h qa1 = load_a16x32(Qbase, DM, q0, 32);

    v8f o0 = {}, o1 = {}, o2 = {}, o3 = {};   // O: 16x64 fp32
    float mrun[8], lrun[8];
#pragma unroll
    for (int r = 0; r < 8; ++r) { mrun[r] = -1e30f; lrun[r] = 0.0f; }

    auto step = [&](int key0) {   // process 32 keys: [key0, key0+32)
        // --- S = Q * K^T  (two 16x16 score tiles) ---
        v8f s0 = {}, s1 = {};
        s0 = wmma_f16(qa0, load_b32x16(Kbase, DM, key0,      0),  s0);
        s0 = wmma_f16(qa1, load_b32x16(Kbase, DM, key0,      32), s0);
        s1 = wmma_f16(qa0, load_b32x16(Kbase, DM, key0 + 16, 0),  s1);
        s1 = wmma_f16(qa1, load_b32x16(Kbase, DM, key0 + 16, 32), s1);

        // --- scale + mask + online softmax ---
        const int c0 = key0 + colIdx;
        const int c1 = key0 + 16 + colIdx;
#pragma unroll
        for (int r = 0; r < 8; ++r) {
            const int q = q0 + r + hi8;
            const bool v0 = ((c0 <= q) && (q - c0 <= WIN)) || (q == 0) || (c0 == 0);
            const bool v1 = ((c1 <= q) && (q - c1 <= WIN)) || (q == 0) || (c1 == 0);
            s0[r] = v0 ? s0[r] * 0.125f : -1e9f;
            s1[r] = v1 ? s1[r] * 0.125f : -1e9f;

            float mt   = redmax16(fmaxf(s0[r], s1[r]));
            float mnew = fmaxf(mrun[r], mt);
            float corr = __expf(mrun[r] - mnew);
            float p0   = __expf(s0[r] - mnew);
            float p1   = __expf(s1[r] - mnew);
            float rs   = redsum16(p0 + p1);
            lrun[r] = lrun[r] * corr + rs;
            mrun[r] = mnew;
            o0[r] *= corr; o1[r] *= corr; o2[r] *= corr; o3[r] *= corr;
            s0[r] = p0; s1[r] = p1;
        }

        // --- P (f32, C-layout) -> LDS f16 -> A-layout ---
        __syncthreads();
#pragma unroll
        for (int r = 0; r < 8; ++r) {
            const int q = r + hi8;
            pst[q * 32 + colIdx]      = (half_t)s0[r];
            pst[q * 32 + 16 + colIdx] = (half_t)s1[r];
        }
        __syncthreads();
        const v16h pa = load_a16x32((const half_t*)pst, 32, 0, 0);

        // --- O += P * V  (V as B-matrix from Vt feature-major layout) ---
        o0 = wmma_f16(pa, load_b32x16(Vbase, NTOK, 0,  key0), o0);
        o1 = wmma_f16(pa, load_b32x16(Vbase, NTOK, 16, key0), o1);
        o2 = wmma_f16(pa, load_b32x16(Vbase, NTOK, 32, key0), o2);
        o3 = wmma_f16(pa, load_b32x16(Vbase, NTOK, 48, key0), o3);
    };

    if (qt == 0) {
        // row 0 is global: attends to every key
        for (int kt2 = 0; kt2 < TSEQ / 16; kt2 += 2) step(kt2 * 16);
    } else {
        int lo = qt - 8; if (lo < 0) lo = 0;       // (q0 - WIN) / 16
        if (lo >= 2) step(0);                      // global column 0 tile-pair
        const int lo2 = lo & ~1;                   // even start; kt2 stays even
        for (int kt2 = lo2; kt2 <= qt; kt2 += 2) step(kt2 * 16);
    }

    // --- normalize and store fp16 attention output [token][d_model] ---
    half_t* Ob = Att + (size_t)(mrow0 + q0) * DM + h * HD;
#pragma unroll
    for (int r = 0; r < 8; ++r) {
        const float inv = 1.0f / lrun[r];          // diagonal always valid -> >0
        const size_t off = (size_t)(r + hi8) * DM + colIdx;
        Ob[off]      = (half_t)(o0[r] * inv);
        Ob[off + 16] = (half_t)(o1[r] * inv);
        Ob[off + 32] = (half_t)(o2[r] * inv);
        Ob[off + 48] = (half_t)(o3[r] * inv);
    }
}

// ---------------------------------------------------------------------------
extern "C" void kernel_launch(void* const* d_in, const int* in_sizes, int n_in,
                              void* d_out, int out_size, void* d_ws, size_t ws_size,
                              hipStream_t stream) {
    const float* x  = (const float*)d_in[0];
    const float* Wq = (const float*)d_in[1];
    const float* bq = (const float*)d_in[2];
    const float* Wk = (const float*)d_in[3];
    const float* bk = (const float*)d_in[4];
    const float* Wv = (const float*)d_in[5];
    const float* bv = (const float*)d_in[6];
    const float* Wo = (const float*)d_in[7];
    const float* bo = (const float*)d_in[8];
    float* out = (float*)d_out;

    const size_t NX = (size_t)NTOK * DM;   // 4M activations
    const size_t NW = (size_t)DM * DM;     // 1M per weight

    half_t* xh  = (half_t*)d_ws;
    half_t* qh  = xh  + NX;
    half_t* kh  = qh  + NX;
    half_t* vt  = kh  + NX;   // V transposed: [DM][NTOK]
    half_t* att = vt  + NX;
    half_t* wqh = att + NX;
    half_t* wkh = wqh + NW;
    half_t* wvh = wkh + NW;
    half_t* woh = wvh + NW;

    // 1) fp32 -> fp16
    f32_to_f16_kernel<<<(int)((NX + 255) / 256), 256, 0, stream>>>(x,  xh,  (int)NX);
    f32_to_f16_kernel<<<(int)((NW + 255) / 256), 256, 0, stream>>>(Wq, wqh, (int)NW);
    f32_to_f16_kernel<<<(int)((NW + 255) / 256), 256, 0, stream>>>(Wk, wkh, (int)NW);
    f32_to_f16_kernel<<<(int)((NW + 255) / 256), 256, 0, stream>>>(Wv, wvh, (int)NW);
    f32_to_f16_kernel<<<(int)((NW + 255) / 256), 256, 0, stream>>>(Wo, woh, (int)NW);

    // 2) QKV projections (V stored transposed)
    dim3 blk(32, 4);
    dim3 grd(DM / 128, NTOK / 32);           // (8, 128)
    gemm_f16out<<<grd, blk, 0, stream>>>(xh, wqh, bq, qh, NTOK, DM, DM, 0);
    gemm_f16out<<<grd, blk, 0, stream>>>(xh, wkh, bk, kh, NTOK, DM, DM, 0);
    gemm_f16out<<<grd, blk, 0, stream>>>(xh, wvh, bv, vt, NTOK, DM, DM, 1);

    // 3) windowed + global flash attention
    attn_kernel<<<dim3(TSEQ / 16, HEADS, BATCH), 32, 0, stream>>>(qh, kh, vt, att);

    // 4) output projection (fp32 result)
    gemm_f32out<<<grd, blk, 0, stream>>>(att, woh, bo, out, NTOK, DM, DM);
}
